// FFOsc_17832704213157
// MI455X (gfx1250) — matrix operand order repaired and checked by reading, not compiled, not run on hardware
//
#include <hip/hip_runtime.h>
#include <hip/hip_bf16.h>
#include <hip/hip_fp16.h>

// ---------------------------------------------------------------------------
// FFOsc for MI455X (gfx1250, wave32, WMMA 16x16x32 f16->f32)
//
// Pipeline:
//   K0 pack_weights  : W_x / W_h  f32 -> f16, pre-swizzled into WMMA B-fragment
//                      lane layout (contiguous v16h per lane at load time)
//   K1 xp_gemm       : xp[c,g,b,t,v] = xs . W_x + b_x   (WMMA, 800 blocks)
//                      stored in C/D-fragment order for coalesced scan reads
//   K2 ff_scan       : 200-step recurrence, 1 workgroup per channel c.
//                      W_h (both gates) staged into 256KB LDS via CDNA5 async
//                      memory->LDS DMA (ASYNCcnt), h in D-layout registers,
//                      double-buffered A-frags, one barrier per step,
//                      16 v_wmma per wave per step.
//   K3 oscillator    : elementwise 10-step oscillator, writes 65MB output.
// ---------------------------------------------------------------------------

typedef __attribute__((ext_vector_type(16))) _Float16 v16h;
typedef __attribute__((ext_vector_type(8)))  float    v8f;

// pointer payload/addrspace types for the async memory->LDS DMA builtin
// (clang-22 prototype: (int4 AS1*, int4 AS3*, imm int, imm int))
typedef int v4i __attribute__((vector_size(16)));
typedef __attribute__((address_space(1))) v4i v4i_glob;
typedef __attribute__((address_space(3))) v4i v4i_lds;

#define OSC_T  10
#define UNITS  256
#define D_INN  256
#define BATCH  32
#define SEQLEN 2000
#define TS     200   // SEQLEN / OSC_T

// workspace layout (bytes)
#define WX_PACK_OFF ((size_t)0)                          // 1 MB  (524288 halves)
#define WH_PACK_OFF ((size_t)1 << 20)                    // 1 MB
#define XP_OFF      ((size_t)2 << 20)                    // 52.4 MB
#define XP_FLOATS   ((size_t)4 * TS * 2 * 8192)          // c,t,g,32tiles*256
#define HS_OFF      (XP_OFF + XP_FLOATS * 4)             // 26.2 MB
// total ~80.7 MB

// CDNA5 async memory->LDS DMA (GLOBAL_LOAD_ASYNC_TO_LDS_B128, ASYNCcnt).
// Guarded with __has_builtin so the compile cannot regress on toolchains
// where the builtin is absent (falls back to VGPR-bounce copy).
#if defined(__gfx1250__) && \
    __has_builtin(__builtin_amdgcn_global_load_async_to_lds_b128) && \
    __has_builtin(__builtin_amdgcn_s_wait_asynccnt)
#define HAVE_ASYNC_LDS 1
#else
#define HAVE_ASYNC_LDS 0
#endif

__device__ __forceinline__ v8f wmma_f16(v16h a, v16h b, v8f c) {
  // D = A(16x32 f16) * B(32x16 f16) + C(16x16 f32)
  return __builtin_amdgcn_wmma_f32_16x16x32_f16(
      /*neg_a=*/false, a, /*neg_b=*/false, b,
      /*c_mod=*/(short)0, c, /*reuse_a=*/false, /*reuse_b=*/false);
}

// half-index inside A-fragment LDS region for element (row M, k K)
// ISA 7.12.2 16-bit A 16x32: lanes0-15 halves = K{0..7,16..23}, lanes16-31 = K{8..15,24..31}
__device__ __forceinline__ int afrag_idx(int row, int k) {
  int mt    = row >> 4;
  int kt    = k >> 5;
  int ko    = k & 31;
  int group = (ko >> 3) & 1;
  int idx   = (ko & 7) + ((ko >> 4) & 1) * 8;
  int lane  = group * 16 + (row & 15);
  return ((mt * 8 + kt) * 32 + lane) * 16 + idx;
}

// ---------------------------------------------------------------------------
// K0: pack a (8 x 256 x 256) f32 weight tensor into B-fragment-ordered f16.
// Packed element tid = ((cg*8+kt)*16+nt)*512 + lane*16 + idx holds
// W[cg][ K = kt*32 + (lane>=16)*8 + (idx&7) + (idx>>3)*16 ][ N = nt*16 + (lane&15) ].
// ---------------------------------------------------------------------------
__global__ void pack_weights(const float* __restrict__ W, _Float16* __restrict__ P) {
  int tid  = blockIdx.x * 256 + threadIdx.x;   // 0 .. 524287
  int idx  = tid & 15;
  int lane = (tid >> 4) & 31;
  int nt   = (tid >> 9) & 15;
  int kt   = (tid >> 13) & 7;
  int cg   = tid >> 16;
  int col  = nt * 16 + (lane & 15);
  int k    = kt * 32 + (lane >> 4) * 8 + (idx & 7) + (idx >> 3) * 16;
  P[tid] = (_Float16)W[((size_t)cg * 256 + k) * 256 + col];
}

// ---------------------------------------------------------------------------
// K1: xp[c,t,g,tile,r,lane] = xs(32x256) . W_x[c,g](256x256) + b_x
// one block per (c,t); 1024 threads = 32 waves; wave -> tile (mt = w>>4, nt = w&15)
// ---------------------------------------------------------------------------
__global__ void __launch_bounds__(1024)
xp_gemm(const float* __restrict__ x, const _Float16* __restrict__ wxp,
        const float* __restrict__ b_x, float* __restrict__ xp) {
  __shared__ __align__(32) _Float16 afrag[8192];   // 16 KB A-fragments (32x256 f16)
  int c = blockIdx.x / TS;
  int t = blockIdx.x % TS;
  int tid = threadIdx.x;

  // stage xs[:, t*OSC_T, :] -> f16 A-fragment layout
  for (int i = 0; i < 8; ++i) {
    int flat = tid + i * 1024;              // 0..8191 = b*256 + d
    int b = flat >> 8, d = flat & 255;
    float v = x[((size_t)b * SEQLEN + (size_t)t * OSC_T) * D_INN + d];
    afrag[afrag_idx(b, d)] = (_Float16)v;
  }
  __syncthreads();

  int wave = tid >> 5, lane = tid & 31;
  int mt = wave >> 4, nt = wave & 15;
  int l  = lane & 15;

  for (int g = 0; g < 2; ++g) {
    int cg = c * 2 + g;
    float bx = b_x[(size_t)cg * UNITS + nt * 16 + l];
    v8f acc;
    for (int r = 0; r < 8; ++r) acc[r] = bx;
    for (int kt = 0; kt < 8; ++kt) {
      v16h a = *(const v16h*)&afrag[((mt * 8 + kt) * 32 + lane) * 16];
      v16h b = *(const v16h*)&wxp[(((size_t)cg * 8 + kt) * 16 + nt) * 512 + lane * 16];
      acc = wmma_f16(a, b, acc);
    }
    // store in C/D-fragment order: coalesced 128B rows, reread coalesced by K2
    size_t base = ((((size_t)c * TS + t) * 2 + g) * 32 + (mt * 16 + nt)) * 256;
    for (int r = 0; r < 8; ++r) xp[base + (size_t)r * 32 + lane] = acc[r];
  }
}

// ---------------------------------------------------------------------------
// K2: the serial recurrence. One block per channel c (4 blocks, 1024 threads).
// W_h (both gates, f16 fragment-packed) resident in 256KB LDS. h lives in
// D-layout registers; rewritten each step into double-buffered A-frag LDS.
// ---------------------------------------------------------------------------
__global__ void __launch_bounds__(1024)
ff_scan(const float* __restrict__ xp, const _Float16* __restrict__ whp,
        const float* __restrict__ b_h, float* __restrict__ hs) {
  __shared__ __align__(32) _Float16 whl[2 * 65536];    // 256 KB: gates 0,1
  __shared__ __align__(32) _Float16 afrag[2][8192];    //  32 KB: double-buffered h

  int c = blockIdx.x;
  int tid = threadIdx.x;
  int wave = tid >> 5, lane = tid & 31;
  int mt = wave >> 4, nt = wave & 15;
  int l  = lane & 15;
  int hi = lane >> 4;

  // stage both gate weight matrices for this channel into LDS (256 KB)
#if HAVE_ASYNC_LDS
  {
    // CDNA5 direct memory->LDS DMA: 16B per lane per op, tracked by ASYNCcnt.
    const char* src = (const char*)(whp + (size_t)c * 131072);
    char* dst = (char*)whl;
    for (int i = tid; i < 16384; i += 1024) {   // 16384 x 16B = 256 KB
      __builtin_amdgcn_global_load_async_to_lds_b128(
          (v4i_glob*)(src + (size_t)i * 16),
          (v4i_lds*)(dst + (size_t)i * 16),
          /*imm offset=*/0, /*cpol=*/0);
    }
    __builtin_amdgcn_s_wait_asynccnt(0);
  }
#else
  {
    const v16h* s = (const v16h*)(whp + (size_t)c * 131072);
    v16h* d = (v16h*)whl;
    for (int i = tid; i < 8192; i += 1024) d[i] = s[i];
  }
#endif
  // h(t=-1) = 0
  for (int i = 0; i < 8; ++i) afrag[0][tid + i * 1024] = (_Float16)0.0f;

  float bh0 = b_h[(size_t)(c * 2 + 0) * UNITS + nt * 16 + l];
  float bh1 = b_h[(size_t)(c * 2 + 1) * UNITS + nt * 16 + l];

  // scatter target for h_new -> A-fragment of next step (u is the K index)
  int u    = nt * 16 + l;
  int kt_a = u >> 5;
  int ko   = u & 31;
  int grp  = (ko >> 3) & 1;
  int sidx = (ko & 7) + ((ko >> 4) & 1) * 8;
  int sbase = ((mt * 8 + kt_a) * 32 + grp * 16 + 8 * hi) * 16 + sidx;

  v8f h;
  for (int r = 0; r < 8; ++r) h[r] = 0.0f;
  __syncthreads();

  for (int t = 0; t < TS; ++t) {
    int buf = t & 1;
    size_t xbase = ((((size_t)c * TS + t) * 2) * 32 + (mt * 16 + nt)) * 256 + lane;

    v8f a0, a1;  // gate accumulators, init = xp_t + b_h
    for (int r = 0; r < 8; ++r) a0[r] = xp[xbase + (size_t)r * 32] + bh0;
    for (int r = 0; r < 8; ++r) a1[r] = xp[xbase + 8192 + (size_t)r * 32] + bh1;

    if (t + 1 < TS) {  // global_prefetch_b8 of next step's xp rows
      __builtin_prefetch(&xp[xbase + 16384], 0, 1);
      __builtin_prefetch(&xp[xbase + 16384 + 8192], 0, 1);
    }

    // hp = h . W_h  (both gates), K = 256 in 8 chunks of 32
    for (int kt = 0; kt < 8; ++kt) {
      v16h a  = *(const v16h*)&afrag[buf][((mt * 8 + kt) * 32 + lane) * 16];
      v16h w0 = *(const v16h*)&whl[((size_t)0 * 65536) + ((kt * 16 + nt) * 32 + lane) * 16];
      v16h w1 = *(const v16h*)&whl[((size_t)1 * 65536) + ((kt * 16 + nt) * 32 + lane) * 16];
      a0 = wmma_f16(a, w0, a0);
      a1 = wmma_f16(a, w1, a1);
    }

    // gates + state update; emit h to hs[t][c][b][u] and to next A-frag buffer
    size_t hbase = (((size_t)t * 4 + c) * 32 + (size_t)(mt * 16 + 8 * hi)) * 256 + u;
    for (int r = 0; r < 8; ++r) {
      float j  = 1.0f / (1.0f + __expf(-a0[r]));
      float k  = 1.0f / (1.0f + __expf(-a1[r]));
      float hn = j + h[r] * (1.0f - j - k);   // j*(1-h) + (1-k)*h
      h[r] = hn;
      hs[hbase + (size_t)r * 256] = hn;
      afrag[buf ^ 1][sbase + (r << 4)] = (_Float16)hn;
    }
    __syncthreads();   // single barrier/step thanks to double buffering
  }
}

// ---------------------------------------------------------------------------
// K3: oscillator. One thread per (t,b,u); 10 serial micro-steps; coalesced out.
// ---------------------------------------------------------------------------
__global__ void oscillator(const float* __restrict__ hs, float* __restrict__ out) {
  int flat = blockIdx.x * 256 + threadIdx.x;     // 0 .. 1638399
  int u = flat & 255;
  int b = (flat >> 8) & 31;
  int t = flat >> 13;
  size_t base = ((size_t)t * 4) * 8192 + (size_t)b * 256 + u;
  float phi = hs[base];            // c = 0
  float om  = hs[base + 8192];     // c = 1
  float r   = hs[base + 16384];    // c = 2
  float mu  = hs[base + 24576];    // c = 3
  size_t obase = ((size_t)b * SEQLEN + (size_t)t * OSC_T) * UNITS + u;
  for (int s = 0; s < OSC_T; ++s) {
    out[obase + (size_t)s * UNITS] = r * __cosf(phi);
    r   = r + (mu - r * r) * r;
    phi = phi + om;
  }
}

// ---------------------------------------------------------------------------
extern "C" void kernel_launch(void* const* d_in, const int* in_sizes, int n_in,
                              void* d_out, int out_size, void* d_ws, size_t ws_size,
                              hipStream_t stream) {
  (void)in_sizes; (void)n_in; (void)out_size; (void)ws_size;
  const float* x   = (const float*)d_in[0];
  const float* W_x = (const float*)d_in[1];
  const float* b_x = (const float*)d_in[2];
  const float* W_h = (const float*)d_in[3];
  const float* b_h = (const float*)d_in[4];
  float* out = (float*)d_out;

  char* ws = (char*)d_ws;
  _Float16* wx_pack = (_Float16*)(ws + WX_PACK_OFF);
  _Float16* wh_pack = (_Float16*)(ws + WH_PACK_OFF);
  float*    xp      = (float*)(ws + XP_OFF);
  float*    hs      = (float*)(ws + HS_OFF);

  pack_weights<<<2048, 256, 0, stream>>>(W_x, wx_pack);
  pack_weights<<<2048, 256, 0, stream>>>(W_h, wh_pack);
  xp_gemm<<<4 * TS, 1024, 0, stream>>>(x, wx_pack, b_x, xp);
  ff_scan<<<4, 1024, 0, stream>>>(xp, wh_pack, b_h, hs);
  oscillator<<<6400, 256, 0, stream>>>(hs, out);
}
